// MatchingPursuit_24455543783756
// MI455X (gfx1250) — matrix operand order repaired and checked by reading, not compiled, not run on hardware
//
#include <hip/hip_runtime.h>
#include <stdint.h>

// ---------------------------------------------------------------------------
// Basis-pursuit ADMM on gfx1250: Woodbury + split-precision bf16 WMMA,
// async-to-LDS double-buffered GEMM pipeline.
//   M^{-1} = I - Wn S^{-1} Wn^T,  S = I_768 + Wn^T Wn  (Newton-Schulz inverse)
// Every Newton-Schulz iterate is a polynomial in symmetric S => symmetric, so
// with Wn^T planes precomputed ALL GEMMs take A as (M x K) and B as (N x K),
// both K-contiguous: staging is pure 16-byte async copies (no LDS transposes).
// ---------------------------------------------------------------------------

#define LAMBD      0.1f
#define ADMM_ITERS 20
#define NS_ITERS   18

typedef __attribute__((ext_vector_type(16))) __bf16         v16bf;
typedef __attribute__((ext_vector_type(8)))  float          v8f;
typedef __attribute__((ext_vector_type(8)))  unsigned short v8us;
typedef __attribute__((ext_vector_type(16))) unsigned short v16us;
typedef int v4i_vs __attribute__((__vector_size__(16)));    // matches builtin param type

union frag16 {
    v16us us;
    v16bf bf;
    v8us  h[2];
};

__device__ __forceinline__ unsigned short f2bf(float f) {
    uint32_t u = __float_as_uint(f);
    uint32_t r = u + 0x7FFFu + ((u >> 16) & 1u);      // round-to-nearest-even
    return (unsigned short)(r >> 16);
}
__device__ __forceinline__ float bf2f(unsigned short h) {
    return __uint_as_float(((uint32_t)h) << 16);
}
__device__ __forceinline__ void split_bf(float f, unsigned short& hi, unsigned short& lo) {
    hi = f2bf(f);
    lo = f2bf(f - bf2f(hi));
}

// ---- CDNA5 async copy global->LDS (ASYNCcnt) --------------------------------
__device__ __forceinline__ void async_b128(const unsigned short* g, unsigned short* l) {
#if __has_builtin(__builtin_amdgcn_global_load_async_to_lds_b128)
    __builtin_amdgcn_global_load_async_to_lds_b128(
        (__attribute__((address_space(1))) v4i_vs*)g,
        (__attribute__((address_space(3))) v4i_vs*)l, 0, 0);
#else
    unsigned ldsa = (unsigned)(uintptr_t)l;   // LDS byte address in VDST
    asm volatile("global_load_async_to_lds_b128 %0, %1, off"
                 :: "v"(ldsa), "v"(g)
                 : "memory");
#endif
}
__device__ __forceinline__ void wait_async0() {
#if __has_builtin(__builtin_amdgcn_s_wait_asynccnt)
    __builtin_amdgcn_s_wait_asynccnt(0);
#else
    asm volatile("s_wait_asynccnt 0" ::: "memory");
#endif
}

// ---------------------------------------------------------------------------
// Split-bf16 WMMA GEMM: C(f32) = A * B^T-view.
//   A planes: M x K row-major. B planes: N x K row-major (K-contiguous).
//   Block tile 64x128, K-tile 32, 128 threads = 4 waves (2x2), wave 32x64.
//   Double-buffered LDS filled by global_load_async_to_lds_b128.
// ---------------------------------------------------------------------------
#define BM 64
#define BN 128
#define BK 32
#define AS (BK + 8)
#define BS (BK + 8)

enum { EPI_STORE = 0, EPI_GRAM = 1, EPI_NS = 2, EPI_ADMM = 3 };

template <int EPI>
__global__ __launch_bounds__(128) void wmma_gemm(
    const unsigned short* __restrict__ Ahi, const unsigned short* __restrict__ Alo, int lda,
    const unsigned short* __restrict__ Bhi, const unsigned short* __restrict__ Blo, int ldb,
    float* __restrict__ Cout, int ldc,
    const float* __restrict__ aux0,   // EPI_NS: X(f32); EPI_ADMM: Adotb
    float* __restrict__ aux1,         // EPI_ADMM: V (read+write)
    float* __restrict__ aux2,         // EPI_ADMM: U (read+write)
    int Kdim)
{
    __shared__ unsigned short shA[2][2][BM * AS];   // [buf][hi/lo][m*AS+k]
    __shared__ unsigned short shB[2][2][BN * BS];   // [buf][hi/lo][n*BS+k]

    const int tid  = threadIdx.x;
    const int lane = tid & 31;
    const int hl   = lane >> 4;        // half-wave
    const int lm   = lane & 15;
    const int wv   = tid >> 5;         // 0..3
    const int wm   = (wv >> 1) * 32;   // wave row base in block tile
    const int wn   = (wv & 1) * 64;    // wave col base in block tile
    const int bm0  = blockIdx.y * BM;
    const int bn0  = blockIdx.x * BN;

    // Issue one K-tile's worth of async copies: 1536 16-byte lane transfers.
    auto stage = [&](int buf, int kt) {
        const int k0 = kt * BK;
        for (int i = tid; i < 1536; i += 128) {
            if (i < 512) {                       // A: 2 planes x 64 rows x 4 chunks
                int pl = i >> 8;
                int r  = (i >> 2) & 63;
                int c  = i & 3;
                const unsigned short* src =
                    (pl ? Alo : Ahi) + (size_t)(bm0 + r) * lda + k0 + c * 8;
                async_b128(src, &shA[buf][pl][r * AS + c * 8]);
            } else {                             // B: 2 planes x 128 rows x 4 chunks
                int j  = i - 512;
                int pl = j >> 9;
                int r  = (j >> 2) & 127;
                int c  = j & 3;
                const unsigned short* src =
                    (pl ? Blo : Bhi) + (size_t)(bn0 + r) * ldb + k0 + c * 8;
                async_b128(src, &shB[buf][pl][r * BS + c * 8]);
            }
        }
    };

    v8f acc[2][4];
    for (int i = 0; i < 2; ++i)
        for (int j = 0; j < 4; ++j)
            for (int e = 0; e < 8; ++e) acc[i][j][e] = 0.0f;

    const int nkt = Kdim / BK;

    stage(0, 0);
    wait_async0();
    __syncthreads();

    for (int kt = 0; kt < nkt; ++kt) {
        const int cur = kt & 1;
        if (kt + 1 < nkt) stage(cur ^ 1, kt + 1);   // overlap with WMMA below

        // ---- fragments per documented 16-bit WMMA layouts ----
        // A 16x32: lanes 0-15 hold K 0..7 & 16..23; lanes 16-31 hold K 8..15 & 24..31.
        frag16 aF[2][2];                            // [plane][mi]
        for (int mi = 0; mi < 2; ++mi) {
            int r = (wm + mi * 16 + lm) * AS + hl * 8;
            for (int pl = 0; pl < 2; ++pl) {
                aF[pl][mi].h[0] = *(const v8us*)&shA[cur][pl][r];
                aF[pl][mi].h[1] = *(const v8us*)&shA[cur][pl][r + 16];
            }
        }
        // B 32x16: lane = N; lanes 0-15 hold K 0..7 & 16..23; lanes 16-31 hold K 8..15 & 24..31.
        frag16 bF[2][4];                            // [plane][ni]
        for (int ni = 0; ni < 4; ++ni) {
            int r = (wn + ni * 16 + lm) * BS + hl * 8;
            for (int pl = 0; pl < 2; ++pl) {
                bF[pl][ni].h[0] = *(const v8us*)&shB[cur][pl][r];
                bF[pl][ni].h[1] = *(const v8us*)&shB[cur][pl][r + 16];
            }
        }

        // ---- split-precision product: hi*hi + hi*lo + lo*hi ----
        for (int mi = 0; mi < 2; ++mi)
            for (int ni = 0; ni < 4; ++ni) {
                acc[mi][ni] = __builtin_amdgcn_wmma_f32_16x16x32_bf16(
                    false, aF[0][mi].bf, false, bF[0][ni].bf, (short)0, acc[mi][ni], false, false);
                acc[mi][ni] = __builtin_amdgcn_wmma_f32_16x16x32_bf16(
                    false, aF[0][mi].bf, false, bF[1][ni].bf, (short)0, acc[mi][ni], false, false);
                acc[mi][ni] = __builtin_amdgcn_wmma_f32_16x16x32_bf16(
                    false, aF[1][mi].bf, false, bF[0][ni].bf, (short)0, acc[mi][ni], false, false);
            }

        wait_async0();        // next tile landed in LDS
        __syncthreads();      // ...for every wave
    }

    // ---- epilogue (C layout: VGPR e -> M = e + 8*half, lane%16 -> N) ----
    for (int mi = 0; mi < 2; ++mi)
        for (int ni = 0; ni < 4; ++ni) {
            int gn = bn0 + wn + ni * 16 + lm;
            int mb = bm0 + wm + mi * 16 + hl * 8;
            for (int e = 0; e < 8; ++e) {
                int gm = mb + e;
                size_t idx = (size_t)gm * ldc + gn;
                float v = acc[mi][ni][e];
                if (EPI == EPI_STORE) {
                    Cout[idx] = v;
                } else if (EPI == EPI_GRAM) {
                    Cout[idx] = v + (gm == gn ? 1.0f : 0.0f);
                } else if (EPI == EPI_NS) {
                    Cout[idx] = 2.0f * aux0[idx] - v;      // Xnew = 2X - X*T
                } else {                                    // fused ADMM update
                    float u   = aux2[idx];
                    float r   = aux0[idx] + aux1[idx] - u;  // adotb + V - U
                    float x   = r - v;                      // x = r - T*Wn^T
                    float xpu = x + u;
                    float mag = fabsf(xpu) - LAMBD;
                    mag = mag > 0.0f ? mag : 0.0f;
                    float vn = (xpu >= 0.0f) ? mag : -mag;  // soft shrink
                    aux1[idx] = vn;                          // V
                    aux2[idx] = xpu - vn;                    // U = u + x - vn
                }
            }
        }
}

// ---------------------------------------------------------------------------
// Elementwise helpers
// ---------------------------------------------------------------------------
__global__ void k_zero(float* __restrict__ p, size_t n) {
    size_t i = (size_t)blockIdx.x * blockDim.x + threadIdx.x;
    size_t st = (size_t)gridDim.x * blockDim.x;
    for (; i < n; i += st) p[i] = 0.0f;
}

__global__ void k_split(const float* __restrict__ s, unsigned short* __restrict__ hi,
                        unsigned short* __restrict__ lo, size_t n) {
    size_t i = (size_t)blockIdx.x * blockDim.x + threadIdx.x;
    size_t st = (size_t)gridDim.x * blockDim.x;
    for (; i < n; i += st) { unsigned short h, l; split_bf(s[i], h, l); hi[i] = h; lo[i] = l; }
}

__global__ void k_prepare(const float* __restrict__ a, const float* __restrict__ v,
                          const float* __restrict__ u, unsigned short* __restrict__ hi,
                          unsigned short* __restrict__ lo, size_t n) {
    size_t i = (size_t)blockIdx.x * blockDim.x + threadIdx.x;
    size_t st = (size_t)gridDim.x * blockDim.x;
    for (; i < n; i += st) {
        unsigned short h, l;
        split_bf(a[i] + v[i] - u[i], h, l);   // R = Adotb + V - U
        hi[i] = h; lo[i] = l;
    }
}

// Normalize dictionary rows; emit Wn planes (mA x nF) AND transposed planes (nF x mA).
__global__ void k_rownorm_split(const float* __restrict__ W,
                                unsigned short* __restrict__ hi, unsigned short* __restrict__ lo,
                                unsigned short* __restrict__ hiT, unsigned short* __restrict__ loT,
                                int nc, int mrows) {
    __shared__ float red[256];
    int row = blockIdx.x;
    float s = 0.0f;
    for (int c = threadIdx.x; c < nc; c += 256) {
        float v = W[(size_t)row * nc + c];
        s += v * v;
    }
    red[threadIdx.x] = s;
    __syncthreads();
    for (int off = 128; off > 0; off >>= 1) {
        if (threadIdx.x < off) red[threadIdx.x] += red[threadIdx.x + off];
        __syncthreads();
    }
    float inv = 1.0f / sqrtf(red[0]);
    for (int c = threadIdx.x; c < nc; c += 256) {
        size_t i = (size_t)row * nc + c;
        unsigned short h, l;
        split_bf(W[i] * inv, h, l);
        hi[i] = h; lo[i] = l;
        size_t it = (size_t)c * mrows + row;
        hiT[it] = h; loT[it] = l;
    }
}

__global__ void k_seti(unsigned int* p) { if (threadIdx.x == 0 && blockIdx.x == 0) *p = 0u; }

__global__ void k_absrowmax(const float* __restrict__ S, int nc, unsigned int* __restrict__ ob) {
    __shared__ float red[256];
    int row = blockIdx.x;
    float s = 0.0f;
    for (int c = threadIdx.x; c < nc; c += 256) s += fabsf(S[(size_t)row * nc + c]);
    red[threadIdx.x] = s;
    __syncthreads();
    for (int off = 128; off > 0; off >>= 1) {
        if (threadIdx.x < off) red[threadIdx.x] += red[threadIdx.x + off];
        __syncthreads();
    }
    if (threadIdx.x == 0) atomicMax(ob, __float_as_uint(red[0]));  // positive floats: bit-monotonic
}

__global__ void k_scale(const float* __restrict__ S, float* __restrict__ X,
                        const unsigned int* __restrict__ bits, size_t n) {
    float s = __uint_as_float(*bits);
    float r = 1.0f / (s * s);               // X0 = S / (||S||_1 ||S||_inf), S symmetric
    size_t i = (size_t)blockIdx.x * blockDim.x + threadIdx.x;
    size_t st = (size_t)gridDim.x * blockDim.x;
    for (; i < n; i += st) X[i] = S[i] * r;
}

// ---------------------------------------------------------------------------
// Host orchestration
// ---------------------------------------------------------------------------
extern "C" void kernel_launch(void* const* d_in, const int* in_sizes, int n_in,
                              void* d_out, int out_size, void* d_ws, size_t ws_size,
                              hipStream_t stream) {
    (void)in_sizes; (void)n_in; (void)out_size; (void)ws_size;

    const float* x  = (const float*)d_in[0];   // (2048,3,16,16) fp32 -> (2048,768)
    const float* wt = (const float*)d_in[1];   // (3072,768) fp32

    const int Bz = 2048, mA = 3072, nF = 768;
    const size_t szMN = (size_t)mA * nF;       // 2359296
    const size_t szBN = (size_t)Bz * nF;       // 1572864
    const size_t szBM = (size_t)Bz * mA;       // 6291456
    const size_t szNN = (size_t)nF * nF;       // 589824

    uintptr_t cur = (uintptr_t)d_ws;
    auto alloc = [&](size_t bytes) -> void* {
        uintptr_t p = (cur + 255) & ~(uintptr_t)255;
        cur = p + bytes;
        return (void*)p;
    };
    unsigned short* WnH  = (unsigned short*)alloc(szMN * 2);   // Wn   (mA x nF)
    unsigned short* WnL  = (unsigned short*)alloc(szMN * 2);
    unsigned short* WnTH = (unsigned short*)alloc(szMN * 2);   // Wn^T (nF x mA)
    unsigned short* WnTL = (unsigned short*)alloc(szMN * 2);
    unsigned short* XfH  = (unsigned short*)alloc(szBN * 2);
    unsigned short* XfL  = (unsigned short*)alloc(szBN * 2);
    float* Adotb         = (float*)alloc(szBM * 4);
    float* V             = (float*)alloc(szBM * 4);
    float* U             = (float*)alloc(szBM * 4);
    unsigned short* RH   = (unsigned short*)alloc(szBM * 2);
    unsigned short* RL   = (unsigned short*)alloc(szBM * 2);
    float* Y             = (float*)alloc(szBN * 4);
    unsigned short* YH   = (unsigned short*)alloc(szBN * 2);
    unsigned short* YL   = (unsigned short*)alloc(szBN * 2);
    float* Tb            = (float*)alloc(szBN * 4);
    unsigned short* TH   = (unsigned short*)alloc(szBN * 2);
    unsigned short* TL   = (unsigned short*)alloc(szBN * 2);
    float* S             = (float*)alloc(szNN * 4);
    unsigned short* SH   = (unsigned short*)alloc(szNN * 2);
    unsigned short* SL   = (unsigned short*)alloc(szNN * 2);
    float* X1            = (float*)alloc(szNN * 4);
    float* X2            = (float*)alloc(szNN * 4);
    float* TNf           = (float*)alloc(szNN * 4);
    unsigned short* XH   = (unsigned short*)alloc(szNN * 2);
    unsigned short* XL   = (unsigned short*)alloc(szNN * 2);
    unsigned short* TNH  = (unsigned short*)alloc(szNN * 2);
    unsigned short* TNL  = (unsigned short*)alloc(szNN * 2);
    unsigned int* scal   = (unsigned int*)alloc(256);

    dim3 blk(256), blkG(128);

    // --- setup: split x, normalize dictionary rows (+ transposed planes) ---
    k_split<<<dim3(2048), blk, 0, stream>>>(x, XfH, XfL, szBN);
    k_rownorm_split<<<dim3(mA), blk, 0, stream>>>(wt, WnH, WnL, WnTH, WnTL, nF, mA);

    // Adotb = xf @ Wn^T : A=(Bz x nF), B rows = Wn rows (mA x nF)
    wmma_gemm<EPI_STORE><<<dim3(mA / BN, Bz / BM), blkG, 0, stream>>>(
        XfH, XfL, nF, WnH, WnL, nF, Adotb, mA, nullptr, nullptr, nullptr, nF);

    // S = Wn^T Wn + I : A = B = Wn^T planes (nF x mA), K = mA
    wmma_gemm<EPI_GRAM><<<dim3(nF / BN, nF / BM), blkG, 0, stream>>>(
        WnTH, WnTL, mA, WnTH, WnTL, mA, S, nF, nullptr, nullptr, nullptr, mA);
    k_split<<<dim3(1024), blk, 0, stream>>>(S, SH, SL, szNN);

    // --- Newton-Schulz inverse of S (all iterates symmetric: N x K == K x N) ---
    k_seti<<<1, 1, 0, stream>>>(scal);
    k_absrowmax<<<dim3(nF), blk, 0, stream>>>(S, nF, scal);
    k_scale<<<dim3(1024), blk, 0, stream>>>(S, X1, scal, szNN);
    float* Xa = X1;
    float* Xb = X2;
    for (int i = 0; i < NS_ITERS; ++i) {
        k_split<<<dim3(1024), blk, 0, stream>>>(Xa, XH, XL, szNN);
        wmma_gemm<EPI_STORE><<<dim3(nF / BN, nF / BM), blkG, 0, stream>>>(
            SH, SL, nF, XH, XL, nF, TNf, nF, nullptr, nullptr, nullptr, nF);
        k_split<<<dim3(1024), blk, 0, stream>>>(TNf, TNH, TNL, szNN);
        wmma_gemm<EPI_NS><<<dim3(nF / BN, nF / BM), blkG, 0, stream>>>(
            XH, XL, nF, TNH, TNL, nF, Xb, nF, Xa, nullptr, nullptr, nF);
        float* t = Xa; Xa = Xb; Xb = t;
    }
    k_split<<<dim3(1024), blk, 0, stream>>>(Xa, XH, XL, szNN);   // XH/XL = S^{-1} planes

    // --- ADMM (fixed 20 iterations == max_iters) ---
    k_zero<<<dim3(4096), blk, 0, stream>>>(V, szBM);
    k_zero<<<dim3(4096), blk, 0, stream>>>(U, szBM);
    for (int it = 0; it < ADMM_ITERS; ++it) {
        k_prepare<<<dim3(4096), blk, 0, stream>>>(Adotb, V, U, RH, RL, szBM);
        // Y = R @ Wn : B rows = Wn^T rows (nF x mA)
        wmma_gemm<EPI_STORE><<<dim3(nF / BN, Bz / BM), blkG, 0, stream>>>(
            RH, RL, mA, WnTH, WnTL, mA, Y, nF, nullptr, nullptr, nullptr, mA);
        k_split<<<dim3(2048), blk, 0, stream>>>(Y, YH, YL, szBN);
        // T = Y @ S^{-1} (symmetric)
        wmma_gemm<EPI_STORE><<<dim3(nF / BN, Bz / BM), blkG, 0, stream>>>(
            YH, YL, nF, XH, XL, nF, Tb, nF, nullptr, nullptr, nullptr, nF);
        k_split<<<dim3(2048), blk, 0, stream>>>(Tb, TH, TL, szBN);
        // fused: x = R - T@Wn^T ; V = shrink(x+U); U = U + x - V   (B rows = Wn rows)
        wmma_gemm<EPI_ADMM><<<dim3(mA / BN, Bz / BM), blkG, 0, stream>>>(
            TH, TL, nF, WnH, WnL, nF, nullptr, mA, Adotb, V, U, nF);
    }

    // --- outputs: z = V, decoded = V @ Wn ---
    float* zout = (float*)d_out;
    (void)hipMemcpyAsync(zout, V, szBM * sizeof(float), hipMemcpyDeviceToDevice, stream);
    k_split<<<dim3(4096), blk, 0, stream>>>(V, RH, RL, szBM);
    wmma_gemm<EPI_STORE><<<dim3(nF / BN, Bz / BM), blkG, 0, stream>>>(
        RH, RL, mA, WnTH, WnTL, mA, zout + szBM, nF, nullptr, nullptr, nullptr, mA);
}
